// RWKV_TimeMix_64450279244534
// MI455X (gfx1250) — matrix operand order repaired and checked by reading, not compiled, not run on hardware
//
#include <hip/hip_runtime.h>
#include <hip/hip_bf16.h>

typedef __attribute__((ext_vector_type(16))) __bf16 v16bf;
typedef __attribute__((ext_vector_type(8)))  __bf16 v8bf;
typedef __attribute__((ext_vector_type(8)))  float  v8f;

#define TT 4096
#define CC 1024          // K == N == 1024
#define MM 16384         // B*T

// ---- gfx1250 async LDS DMA path (guarded; falls back to sync copy) --------
#if defined(__has_builtin)
#if __has_builtin(__builtin_amdgcn_global_load_async_to_lds_b128) && \
    __has_builtin(__builtin_amdgcn_s_wait_asynccnt)
#define USE_ASYNC_LDS 1
#endif
#endif
#ifndef USE_ASYNC_LDS
#define USE_ASYNC_LDS 0
#endif

typedef int v4i_t __attribute__((vector_size(16)));
typedef __attribute__((address_space(1))) v4i_t glb_v4i;
typedef __attribute__((address_space(3))) v4i_t lds_v4i;

__device__ __forceinline__ lds_v4i* to_lds(void* p) {
    // generic LDS address carries the LDS byte offset in its low 32 bits
    return (lds_v4i*)(unsigned int)(unsigned long long)p;
}
__device__ __forceinline__ glb_v4i* to_glb(const void* p) {
    return (glb_v4i*)(unsigned long long)p;
}

// ---------------------------------------------------------------------------
// Prep 1: fused time-shift mix, f32 -> bf16.  8 elements / thread.
// ---------------------------------------------------------------------------
__global__ __launch_bounds__(256)
void prep_mix(const float* __restrict__ x,
              const float* __restrict__ tmk, const float* __restrict__ tmv,
              const float* __restrict__ tmr,
              __bf16* __restrict__ xk, __bf16* __restrict__ xv,
              __bf16* __restrict__ xr)
{
    const int    gid = blockIdx.x * 256 + threadIdx.x;
    const size_t e   = (size_t)gid * 8;
    const int    m   = (int)(e >> 10);
    const bool   t0  = (m & (TT - 1)) == 0;
    const int    kk  = (int)(e & (CC - 1));

    float xa[8], pa[8];
    {
        const float4* xp = (const float4*)(x + e);
        float4 c0 = xp[0], c1 = xp[1];
        xa[0]=c0.x; xa[1]=c0.y; xa[2]=c0.z; xa[3]=c0.w;
        xa[4]=c1.x; xa[5]=c1.y; xa[6]=c1.z; xa[7]=c1.w;
        if (t0) {
            #pragma unroll
            for (int i = 0; i < 8; ++i) pa[i] = 0.0f;
        } else {
            const float4* pp = (const float4*)(x + e - CC);
            float4 p0 = pp[0], p1 = pp[1];
            pa[0]=p0.x; pa[1]=p0.y; pa[2]=p0.z; pa[3]=p0.w;
            pa[4]=p1.x; pa[5]=p1.y; pa[6]=p1.z; pa[7]=p1.w;
        }
    }
    const float* mixes[3] = { tmk, tmv, tmr };
    __bf16*      outs [3] = { xk,  xv,  xr  };
    #pragma unroll
    for (int s = 0; s < 3; ++s) {
        const float4* mp = (const float4*)(mixes[s] + kk);
        float4 m0 = mp[0], m1 = mp[1];
        float mm[8] = { m0.x,m0.y,m0.z,m0.w, m1.x,m1.y,m1.z,m1.w };
        v8bf o;
        #pragma unroll
        for (int i = 0; i < 8; ++i)
            o[i] = (__bf16)(xa[i] * mm[i] + pa[i] * (1.0f - mm[i]));
        *(v8bf*)(outs[s] + e) = o;
    }
}

// ---------------------------------------------------------------------------
// Prep 2: weight f32 -> bf16 (8 elements / thread)
// ---------------------------------------------------------------------------
__global__ __launch_bounds__(256)
void prep_w(const float* __restrict__ src, __bf16* __restrict__ dst)
{
    const size_t e = (size_t)(blockIdx.x * 256 + threadIdx.x) * 8;
    const float4* sp = (const float4*)(src + e);
    float4 c0 = sp[0], c1 = sp[1];
    float a[8] = { c0.x,c0.y,c0.z,c0.w, c1.x,c1.y,c1.z,c1.w };
    v8bf o;
    #pragma unroll
    for (int i = 0; i < 8; ++i) o[i] = (__bf16)a[i];
    *(v8bf*)(dst + e) = o;
}

// ---------------------------------------------------------------------------
// Y[M,N](f32) = A[M,K](bf16) * W[N,K](bf16)^T
// Block: 256 threads (8 wave32s). Tile 128x128, K-step 32.
// Double-buffered LDS fed by GLOBAL_LOAD_ASYNC_TO_LDS_B128 (ASYNCcnt).
// ---------------------------------------------------------------------------
__global__ __launch_bounds__(256)
void gemm_bf16_wmma(const __bf16* __restrict__ A, const __bf16* __restrict__ W,
                    float* __restrict__ Y)
{
    __shared__ __align__(32) __bf16 As[2][128 * 32];
    __shared__ __align__(32) __bf16 Bs[2][128 * 32];

    const int nblk = blockIdx.x;     // 0..7
    const int mblk = blockIdx.y;     // 0..127
    const int tid  = threadIdx.x;
    const int wave = tid >> 5;
    const int lane = tid & 31;

    v8f acc[8];
    #pragma unroll
    for (int i = 0; i < 8; ++i)
        #pragma unroll
        for (int j = 0; j < 8; ++j) acc[i][j] = 0.0f;

    // Per tile: 128 rows x 32 bf16 = 128 x 64B = 512 b128 chunks for A and B.
    // Each thread moves 2 A-chunks + 2 B-chunks.
    auto issue_tile = [&](int buf, int kb) {
        #pragma unroll
        for (int i = 0; i < 2; ++i) {
            const int c   = tid + i * 256;      // 0..511
            const int row = c >> 2;             // 0..127
            const int q   = c & 3;              // 16B quarter
            const __bf16* ga = A + (size_t)(mblk * 128 + row) * CC + kb + q * 8;
            const __bf16* gb = W + (size_t)(nblk * 128 + row) * CC + kb + q * 8;
            __bf16* la = &As[buf][row * 32 + q * 8];
            __bf16* lb = &Bs[buf][row * 32 + q * 8];
#if USE_ASYNC_LDS
            __builtin_amdgcn_global_load_async_to_lds_b128(to_glb(ga), to_lds(la), 0, 0);
            __builtin_amdgcn_global_load_async_to_lds_b128(to_glb(gb), to_lds(lb), 0, 0);
#else
            *(uint4*)la = *(const uint4*)ga;
            *(uint4*)lb = *(const uint4*)gb;
#endif
        }
    };

    issue_tile(0, 0);
#if USE_ASYNC_LDS
    __builtin_amdgcn_s_wait_asynccnt(0);
#endif
    __syncthreads();

    for (int kb = 0; kb < CC; kb += 32) {
        const int buf = (kb >> 5) & 1;
        if (kb + 32 < CC) issue_tile(buf ^ 1, kb + 32);

        // A fragment (16x32 bf16), contiguous K order matching B:
        //   lanes 0-15 : row = lane,    K 0..15
        //   lanes 16-31: row = lane-16, K 16..31
        const int arow = wave * 16 + (lane & 15);
        const int ak0  = (lane >> 4) * 16;
        v8bf alo = *(const v8bf*)&As[buf][arow * 32 + ak0];
        v8bf ahi = *(const v8bf*)&As[buf][arow * 32 + ak0 + 8];
        v16bf afrag = __builtin_shufflevector(alo, ahi,
                        0,1,2,3,4,5,6,7,8,9,10,11,12,13,14,15);

        #pragma unroll
        for (int nt = 0; nt < 8; ++nt) {
            const int bcol = nt * 16 + (lane & 15);
            const int bk0  = (lane >> 4) * 16;
            v16bf bfrag = *(const v16bf*)&Bs[buf][bcol * 32 + bk0];
            acc[nt] = __builtin_amdgcn_wmma_f32_16x16x32_bf16(
                          false, afrag, false, bfrag,
                          (short)0, acc[nt], false, false);
        }

#if USE_ASYNC_LDS
        __builtin_amdgcn_s_wait_asynccnt(0);
#endif
        __syncthreads();
    }

    // D layout: lanes 0-15 -> M=r, lanes 16-31 -> M=8+r; N = lane&15
    const int orow0 = mblk * 128 + wave * 16 + ((lane >> 4) << 3);
    const int ocol0 = nblk * 128 + (lane & 15);
    #pragma unroll
    for (int nt = 0; nt < 8; ++nt)
        #pragma unroll
        for (int r = 0; r < 8; ++r)
            Y[(size_t)(orow0 + r) * CC + ocol0 + nt * 16] = acc[nt][r];
}

// ---------------------------------------------------------------------------
// WKV recurrence: one thread per (b, c) channel, sequential over T, with a
// one-step load prefetch so k/v/r for t+1 are in flight during the exp chain.
// Fuses sigmoid(r) * y and emits bf16 for the output GEMM.
// ---------------------------------------------------------------------------
__global__ __launch_bounds__(256)
void wkv_scan(const float* __restrict__ k, const float* __restrict__ v,
              const float* __restrict__ r,
              const float* __restrict__ time_decay,
              const float* __restrict__ time_first,
              __bf16* __restrict__ y)
{
    const int gid = blockIdx.x * blockDim.x + threadIdx.x;   // 0..B*C-1
    const int b = gid / CC;
    const int c = gid - b * CC;

    const float w = -__expf(time_decay[c]);
    const float u = time_first[c];

    float aa = 0.0f, bb = 0.0f, pp = -1e38f;
    size_t idx = (size_t)b * TT * CC + c;

    float kt = k[idx], vt = v[idx], rt = r[idx];
    for (int t = 0; t < TT; ++t) {
        const size_t nidx = idx + CC;
        float kn = 0.0f, vn = 0.0f, rn = 0.0f;
        if (t + 1 < TT) { kn = k[nidx]; vn = v[nidx]; rn = r[nidx]; }

        const float ww = u + kt;
        const float p  = fmaxf(pp, ww);
        const float e1 = __expf(pp - p);
        const float e2 = __expf(ww - p);
        const float yt = (e1 * aa + e2 * vt) / (e1 * bb + e2);

        const float sr = 1.0f / (1.0f + __expf(-rt));
        y[idx] = (__bf16)(sr * yt);

        const float ww2 = pp + w;
        const float p2  = fmaxf(ww2, kt);
        const float e1b = __expf(ww2 - p2);
        const float e2b = __expf(kt - p2);
        aa = e1b * aa + e2b * vt;
        bb = e1b * bb + e2b;
        pp = p2;

        kt = kn; vt = vn; rt = rn; idx = nidx;
    }
}

// ---------------------------------------------------------------------------
extern "C" void kernel_launch(void* const* d_in, const int* in_sizes, int n_in,
                              void* d_out, int out_size, void* d_ws, size_t ws_size,
                              hipStream_t stream)
{
    const float* x   = (const float*)d_in[0];
    const float* td  = (const float*)d_in[1];  // time_decay
    const float* tf  = (const float*)d_in[2];  // time_first
    const float* tmk = (const float*)d_in[3];
    const float* tmv = (const float*)d_in[4];
    const float* tmr = (const float*)d_in[5];
    const float* Wk  = (const float*)d_in[6];
    const float* Wv  = (const float*)d_in[7];
    const float* Wr  = (const float*)d_in[8];
    const float* Wo  = (const float*)d_in[9];
    float* out = (float*)d_out;

    const size_t SZ  = (size_t)MM * CC;        // 16M activations
    const size_t WSZ = (size_t)CC * CC;        // 1M weights

    __bf16* xkbf = (__bf16*)d_ws;
    __bf16* xvbf = xkbf + SZ;
    __bf16* xrbf = xvbf + SZ;
    __bf16* ybf  = xrbf + SZ;
    __bf16* wkbf = ybf  + SZ;
    __bf16* wvbf = wkbf + WSZ;
    __bf16* wrbf = wvbf + WSZ;
    __bf16* wobf = wrbf + WSZ;
    float*  kf   = (float*)(wobf + WSZ);
    float*  vf   = kf + SZ;
    float*  rf   = vf + SZ;

    dim3 blk(256);

    prep_mix<<<dim3((unsigned)(SZ / 8 / 256)), blk, 0, stream>>>(
        x, tmk, tmv, tmr, xkbf, xvbf, xrbf);
    prep_w<<<dim3((unsigned)(WSZ / 8 / 256)), blk, 0, stream>>>(Wk, wkbf);
    prep_w<<<dim3((unsigned)(WSZ / 8 / 256)), blk, 0, stream>>>(Wv, wvbf);
    prep_w<<<dim3((unsigned)(WSZ / 8 / 256)), blk, 0, stream>>>(Wr, wrbf);
    prep_w<<<dim3((unsigned)(WSZ / 8 / 256)), blk, 0, stream>>>(Wo, wobf);

    dim3 grid(CC / 128, MM / 128);             // (8, 128)
    gemm_bf16_wmma<<<grid, blk, 0, stream>>>(xkbf, wkbf, kf);
    gemm_bf16_wmma<<<grid, blk, 0, stream>>>(xvbf, wvbf, vf);
    gemm_bf16_wmma<<<grid, blk, 0, stream>>>(xrbf, wrbf, rf);

    wkv_scan<<<dim3((4 * CC) / 256), blk, 0, stream>>>(kf, vf, rf, td, tf, ybf);

    gemm_bf16_wmma<<<grid, blk, 0, stream>>>(ybf, wobf, out);
}
